// FractalCompressor_46961172414852
// MI455X (gfx1250) — matrix operand order repaired and checked by reading, not compiled, not run on hardware
//
#include <hip/hip_runtime.h>
#include <hip/hip_bf16.h>

typedef __attribute__((ext_vector_type(16))) __bf16       v16bf;
typedef __attribute__((ext_vector_type(8)))  __bf16       v8bf;
typedef __attribute__((ext_vector_type(8)))  float        v8f;
typedef __attribute__((ext_vector_type(4)))  unsigned int v4u;

#define DEV __device__ __forceinline__

#define BATCH 8
#define SEQ   2047
#define DIM   512
#define HH    256

DEV v8f splat8(float x) { v8f r = {x,x,x,x,x,x,x,x}; return r; }

// Register pass-through: pins a loaded value as an asm-defined register so
// consumers (WMMA) cannot be hoisted above this point, forcing batched loads.
DEV void touch4(v4u& a) { asm volatile("" : "+v"(a)); }

DEV v16bf join16(v4u lo4, v4u hi4) {
    v8bf lo = __builtin_bit_cast(v8bf, lo4);
    v8bf hi = __builtin_bit_cast(v8bf, hi4);
    return __builtin_shufflevector(lo, hi, 0,1,2,3,4,5,6,7,8,9,10,11,12,13,14,15);
}

// Build a 16x32 bf16 A-tile operand in contiguous k order (same as B):
// lanes 0-15 hold K[0..15]; lanes 16-31 hold K[16..31].
DEV v16bf make_a(const __bf16* row, int off0, int off1, bool valid) {
    v4u z = {0u,0u,0u,0u};
    v4u q0 = valid ? *(const v4u*)(row + off0) : z;
    v4u q1 = valid ? *(const v4u*)(row + off1) : z;
    return join16(q0, q1);
}

DEV v8f wmma_bf16(v16bf a, v16bf b, v8f c) {
    return __builtin_amdgcn_wmma_f32_16x16x32_bf16(false, a, false, b, (short)0, c, false, false);
}

// CDNA5 async DMA: global -> LDS, tracked by ASYNCcnt (no VGPR round-trip).
DEV void async_lds_b128(unsigned lds_byte_off, const __bf16* g) {
    asm volatile("global_load_async_to_lds_b128 %0, %1, off"
                 :: "v"(lds_byte_off), "v"(g) : "memory");
}
DEV void wait_async0() { asm volatile("s_wait_asynccnt 0" ::: "memory"); }

DEV float sigm(float x)  { return 1.0f / (1.0f + __expf(-x)); }
DEV float tanhx(float x) { return 1.0f - 2.0f / (1.0f + __expf(2.0f * x)); }

// ---------------- prep kernels ----------------

__global__ void cvt_bf16_kernel(const float* __restrict__ src, __bf16* __restrict__ dst, int n) {
    int i = blockIdx.x * blockDim.x + threadIdx.x;
    int stride = gridDim.x * blockDim.x;
    for (; i < n; i += stride) dst[i] = (__bf16)src[i];
}

// w[256][512][k]  ->  wp[k][256][512]  (bf16): B-tiles become K-contiguous per column.
__global__ void pack_convw_kernel(const float* __restrict__ w, __bf16* __restrict__ wp, int k) {
    int n = 256 * 512 * k;
    for (int i = blockIdx.x * blockDim.x + threadIdx.x; i < n; i += gridDim.x * blockDim.x) {
        int t = i / (256 * 512);
        int rem = i - t * (256 * 512);
        int c = rem >> 9;
        int d = rem & 511;
        wp[i] = (__bf16)w[(c * 512 + d) * k + t];
    }
}

__global__ void bias_sum_kernel(const float* __restrict__ a, const float* __restrict__ b,
                                float* __restrict__ o, int n) {
    int i = blockIdx.x * blockDim.x + threadIdx.x;
    if (i < n) o[i] = a[i] + b[i];
}

// ---------------- fused conv + bias + relu + pool(8) ----------------
// Grid: 4 scales * 8 batch * 16 mtiles = 512 blocks, 256 threads (8 waves).
// WG tile: M=128 conv outputs (16 pooled j), N=256 channels. Wave w: rows [w*16, w*16+16).
// B (weights) staged in double-buffered LDS by the async DMA engine; all 8 waves share it.
#define BSTRIDE 40   // bf16 elems per staged column (80B): 16 lanes -> 16 distinct banks

__global__ __launch_bounds__(256, 2)
void conv_pool_kernel(const __bf16* __restrict__ xb, const __bf16* __restrict__ wp,
                      const float* __restrict__ b1, const float* __restrict__ b2,
                      const float* __restrict__ b3, const float* __restrict__ b4,
                      __bf16* __restrict__ seqb) {
    __shared__ __bf16 Bs[2 * 256 * BSTRIDE];   // 2 x 20KB

    int bx = blockIdx.x;
    int scale = bx >> 7;
    int b     = (bx >> 4) & 7;
    int mt    = bx & 15;
    int k   = 2 << scale;
    int pad = k >> 1;
    const int wpoff[4] = {0, 262144, 786432, 1835008};
    const __bf16* wps = wp + wpoff[scale];
    const float* bias = (scale == 0) ? b1 : (scale == 1) ? b2 : (scale == 2) ? b3 : b4;

    int tid  = threadIdx.x;
    int wv   = tid >> 5;
    int lane = tid & 31;
    int half = lane >> 4;
    int lr   = lane & 15;
    int m0   = mt * 128 + wv * 16;

    // async fill decomposition: thread covers cols c0+64j (j=0..3), 16B sub-chunk 'sub'
    int c0  = tid >> 2;
    int sub = tid & 3;
    unsigned lds_base = (unsigned)(uintptr_t)(&Bs[0]);

    v8f acc[16];
#pragma unroll
    for (int i = 0; i < 16; ++i) acc[i] = splat8(0.0f);

    const __bf16* xrow_base = xb + (size_t)b * SEQ * DIM;
    int nk = k * 16;   // ksteps: ks = t*16 + d32

    // prologue: fill buffer 0 for ks=0
    {
        const __bf16* gsrc = wps + /*t=0, kb=0*/ sub * 8;
#pragma unroll
        for (int j = 0; j < 4; ++j) {
            int col = c0 + 64 * j;
            async_lds_b128(lds_base + (unsigned)(col * (BSTRIDE * 2) + sub * 16),
                           gsrc + (size_t)col * DIM);
        }
    }
    wait_async0();
    __syncthreads();

    for (int ks = 0; ks < nk; ++ks) {
        int cur = ks & 1;
        // kick off async DMA of next kstep's B chunk into the other buffer
        if (ks + 1 < nk) {
            int tn = (ks + 1) >> 4;
            int kbn = ((ks + 1) & 15) * 32;
            const __bf16* gsrc = wps + (size_t)tn * (HH * DIM) + kbn + sub * 8;
            unsigned dst = lds_base + (unsigned)((cur ^ 1) * (256 * BSTRIDE * 2) + sub * 16);
#pragma unroll
            for (int j = 0; j < 4; ++j) {
                int col = c0 + 64 * j;
                async_lds_b128(dst + (unsigned)(col * (BSTRIDE * 2)), gsrc + (size_t)col * DIM);
            }
        }

        int t  = ks >> 4;
        int kb = (ks & 15) * 32;
        int row = m0 + lr - pad + t;
        bool valid = (row >= 0) && (row < SEQ);
        int rowc = valid ? row : 0;
        v16bf A = make_a(xrow_base + (size_t)rowc * DIM, kb + half * 16, kb + half * 16 + 8, valid);

        // Batch all 32 ds_load_b128 for the 16 B fragments; the register pass-through
        // pins them so WMMAs cannot hoist between load pairs -> one LDS latency/kstep.
        const __bf16* bbase = &Bs[cur * (256 * BSTRIDE)];
        v4u qlo[16], qhi[16];
#pragma unroll
        for (int nt = 0; nt < 16; ++nt) {
            const __bf16* p = bbase + (nt * 16 + lr) * BSTRIDE + half * 16;
            qlo[nt] = *(const v4u*)p;
            qhi[nt] = *(const v4u*)(p + 8);
        }
#pragma unroll
        for (int nt = 0; nt < 16; ++nt) { touch4(qlo[nt]); touch4(qhi[nt]); }
#pragma unroll
        for (int nt = 0; nt < 16; ++nt)
            acc[nt] = wmma_bf16(A, join16(qlo[nt], qhi[nt]), acc[nt]);

        wait_async0();      // next buffer's DMA done (overlapped with the 16 WMMAs above)
        __syncthreads();    // all waves done reading 'cur' + all fills visible
    }

    // bias + relu + mean-of-8 pool: each lane's 8 accum VGPRs are exactly one pool window.
    int j = (m0 >> 3) + half;
    __bf16* outp = seqb + (((size_t)scale * BATCH + b) * HH + j) * HH;
#pragma unroll
    for (int nt = 0; nt < 16; ++nt) {
        int ch = nt * 16 + lr;
        float bsv = bias[ch];
        float s = 0.0f;
#pragma unroll
        for (int i = 0; i < 8; ++i) s += fmaxf(acc[nt][i] + bsv, 0.0f);
        outp[ch] = (__bf16)(s * 0.125f);
    }
}

// ---------------- LSTM input projection: Gin[scale][t][g][b] = bsum[g] + seq . wih^T ----------------
__global__ __launch_bounds__(256, 2)
void proj_kernel(const __bf16* __restrict__ seqb, const __bf16* __restrict__ wihb,
                 const float* __restrict__ bsum, float* __restrict__ gin) {
    int bx = blockIdx.x;
    int scale = bx >> 7;
    int b     = (bx >> 4) & 7;
    int mt    = (bx >> 2) & 3;
    int nb    = bx & 3;
    int wv = threadIdx.x >> 5, lane = threadIdx.x & 31, half = lane >> 4, lr = lane & 15;
    int wm  = wv >> 1;
    int ntb = (wv & 1) * 8;
    int t0  = mt * 64 + wm * 16;
    const __bf16* arow0 = seqb + (((size_t)scale * BATCH + b) * HH + t0) * HH;

    v8f acc[8];
#pragma unroll
    for (int nt = 0; nt < 8; ++nt) {
        int g = nb * 256 + (ntb + nt) * 16 + lr;
        acc[nt] = splat8(bsum[g]);
    }
    for (int ks = 0; ks < 8; ++ks) {
        int kb = ks * 32;
        v16bf A = make_a(arow0 + (size_t)lr * HH, kb + half * 16, kb + half * 16 + 8, true);
        v4u qlo[8], qhi[8];
#pragma unroll
        for (int nt = 0; nt < 8; ++nt) {
            int g = nb * 256 + (ntb + nt) * 16 + lr;
            const __bf16* p = wihb + (size_t)g * HH + kb + half * 16;
            qlo[nt] = *(const v4u*)p;
            qhi[nt] = *(const v4u*)(p + 8);
        }
#pragma unroll
        for (int nt = 0; nt < 8; ++nt) { touch4(qlo[nt]); touch4(qhi[nt]); }
#pragma unroll
        for (int nt = 0; nt < 8; ++nt)
            acc[nt] = wmma_bf16(A, join16(qlo[nt], qhi[nt]), acc[nt]);
    }
#pragma unroll
    for (int nt = 0; nt < 8; ++nt) {
        int g = nb * 256 + (ntb + nt) * 16 + lr;
#pragma unroll
        for (int i = 0; i < 8; ++i) {
            int t = t0 + half * 8 + i;
            gin[(((size_t)scale * 256 + t) * 1024 + g) * 8 + b] = acc[nt][i];
        }
    }
}

// ---------------- recurrent LSTM + output mean over t ----------------
// Grid: 4 blocks (one per scale), 256 threads. Wave w owns n-tiles {w+8q}: g = 16w+lr+128q,
// so q=0,1->i | 2,3->f | 4,5->g | 6,7->o for hidx = 16w+lr+128*q0 -> gate fusion is register-local.
__global__ __launch_bounds__(256, 2)
void lstm_kernel(const float* __restrict__ gin, const __bf16* __restrict__ whhb,
                 float* __restrict__ out) {
    int scale = blockIdx.x;
    int wv = threadIdx.x >> 5, lane = threadIdx.x & 31, half = lane >> 4, lr = lane & 15;
    __shared__ __bf16 hb[16 * 256];

    {
        v4u z = {0u,0u,0u,0u};
        for (int i = threadIdx.x; i < (16 * 256) / 8; i += 256) ((v4u*)hb)[i] = z;
    }
    __syncthreads();

    v8f cst[2], hsum[2];
#pragma unroll
    for (int q0 = 0; q0 < 2; ++q0) { cst[q0] = splat8(0.0f); hsum[q0] = splat8(0.0f); }

    const float* ginS = gin + (size_t)scale * 256 * 1024 * 8;

    for (int t = 0; t < 256; ++t) {
        const float* gt = ginS + (size_t)t * 1024 * 8;
        v8f acc[8];
#pragma unroll
        for (int q = 0; q < 8; ++q) {
            int g = 16 * wv + lr + 128 * q;
            v8f z = splat8(0.0f);
            if (half == 0) z = *(const v8f*)(gt + (size_t)g * 8);  // batches 0..7 = C rows 0..7
            acc[q] = z;                                            // rows 8..15: zero padding
        }
        for (int ks = 0; ks < 8; ++ks) {
            int kb = ks * 32;
            v16bf A = make_a(hb + lr * 256, kb + half * 16, kb + half * 16 + 8, true);
            v4u qlo[8], qhi[8];
#pragma unroll
            for (int q = 0; q < 8; ++q) {
                int g = 16 * wv + lr + 128 * q;
                const __bf16* p = whhb + (size_t)g * HH + kb + half * 16;
                qlo[q] = *(const v4u*)p;
                qhi[q] = *(const v4u*)(p + 8);
            }
#pragma unroll
            for (int q = 0; q < 8; ++q) { touch4(qlo[q]); touch4(qhi[q]); }
#pragma unroll
            for (int q = 0; q < 8; ++q)
                acc[q] = wmma_bf16(A, join16(qlo[q], qhi[q]), acc[q]);
        }
        __syncthreads();   // all waves done reading hb for this step
        if (half == 0) {
#pragma unroll
            for (int q0 = 0; q0 < 2; ++q0) {
                int hidx = 16 * wv + lr + 128 * q0;
#pragma unroll
                for (int i = 0; i < 8; ++i) {
                    float iv = sigm(acc[q0][i]);
                    float fv = sigm(acc[q0 + 2][i]);
                    float gv = tanhx(acc[q0 + 4][i]);
                    float ov = sigm(acc[q0 + 6][i]);
                    float c  = fv * cst[q0][i] + iv * gv;
                    cst[q0][i] = c;
                    float h = ov * tanhx(c);
                    hsum[q0][i] += h;
                    hb[i * 256 + hidx] = (__bf16)h;
                }
            }
        }
        __syncthreads();   // hb updated for next step
    }
    if (half == 0) {
#pragma unroll
        for (int q0 = 0; q0 < 2; ++q0) {
            int hidx = 16 * wv + lr + 128 * q0;
#pragma unroll
            for (int i = 0; i < 8; ++i)
                out[(size_t)i * 1024 + scale * 256 + hidx] = hsum[q0][i] * (1.0f / 256.0f);
        }
    }
}

// ---------------- host launch ----------------

extern "C" void kernel_launch(void* const* d_in, const int* in_sizes, int n_in,
                              void* d_out, int out_size, void* d_ws, size_t ws_size,
                              hipStream_t stream) {
    const float* x    = (const float*)d_in[0];
    const float* cw[4] = {(const float*)d_in[1], (const float*)d_in[3],
                          (const float*)d_in[5], (const float*)d_in[7]};
    const float* cb1 = (const float*)d_in[2];
    const float* cb2 = (const float*)d_in[4];
    const float* cb3 = (const float*)d_in[6];
    const float* cb4 = (const float*)d_in[8];
    const float* wih = (const float*)d_in[9];
    const float* whh = (const float*)d_in[10];
    const float* bih = (const float*)d_in[11];
    const float* bhh = (const float*)d_in[12];
    float* out = (float*)d_out;

    // workspace carve (segments 256B aligned)
    char* base = (char*)d_ws;
    const size_t off_xb   = 0;                              // 8*2047*512 bf16 = 16,769,024 B
    const size_t off_wp   = off_xb  + 16769024;             // packed conv weights: 7,864,320 B
    const size_t off_wih  = off_wp  + 7864320;              // 1024*256 bf16 = 524,288 B
    const size_t off_whh  = off_wih + 524288;
    const size_t off_bs   = off_whh + 524288;               // 1024 f32 = 4,096 B
    const size_t off_seqb = off_bs  + 4096;                 // 4*8*256*256 bf16 = 4,194,304 B
    const size_t off_gin  = off_seqb + 4194304;             // 4*256*1024*8 f32 = 33,554,432 B

    __bf16* xb   = (__bf16*)(base + off_xb);
    __bf16* wp   = (__bf16*)(base + off_wp);
    __bf16* wihb = (__bf16*)(base + off_wih);
    __bf16* whhb = (__bf16*)(base + off_whh);
    float*  bs   = (float*)(base + off_bs);
    __bf16* seqb = (__bf16*)(base + off_seqb);
    float*  gin  = (float*)(base + off_gin);

    // 1) x -> bf16
    cvt_bf16_kernel<<<4096, 256, 0, stream>>>(x, xb, BATCH * SEQ * DIM);

    // 2) conv weights -> [t][c][d] bf16
    const int wpoff[4] = {0, 262144, 786432, 1835008};
    for (int s = 0; s < 4; ++s) {
        int k = 2 << s;
        int n = 256 * 512 * k;
        int blocks = (n + 255) / 256; if (blocks > 2048) blocks = 2048;
        pack_convw_kernel<<<blocks, 256, 0, stream>>>(cw[s], wp + wpoff[s], k);
    }

    // 3) LSTM weights -> bf16, bias presum
    cvt_bf16_kernel<<<1024, 256, 0, stream>>>(wih, wihb, 1024 * 256);
    cvt_bf16_kernel<<<1024, 256, 0, stream>>>(whh, whhb, 1024 * 256);
    bias_sum_kernel<<<4, 256, 0, stream>>>(bih, bhh, bs, 1024);

    // 4) fused conv + bias + relu + pool -> seqb
    conv_pool_kernel<<<512, 256, 0, stream>>>(xb, wp, cb1, cb2, cb3, cb4, seqb);

    // 5) input projection -> gin
    proj_kernel<<<512, 256, 0, stream>>>(seqb, wihb, bs, gin);

    // 6) recurrent LSTM + temporal mean -> out
    lstm_kernel<<<4, 256, 0, stream>>>(gin, whhb, out);

    (void)in_sizes; (void)n_in; (void)out_size; (void)ws_size;
}
